// GaussianMixtureModel_77446850282210
// MI455X (gfx1250) — matrix operand order repaired and checked by reading, not compiled, not run on hardware
//
#include <hip/hip_runtime.h>
#include <hip/hip_bf16.h>
#include <math.h>

#define NPTS 8192
#define KCMP 64
#define DDIM 256

typedef __bf16 bf16_t;
typedef bf16_t bf16x8 __attribute__((ext_vector_type(8)));
typedef bf16_t v16bf  __attribute__((ext_vector_type(16)));
typedef float  v8f    __attribute__((ext_vector_type(8)));
typedef unsigned int u32x4 __attribute__((ext_vector_type(4)));
typedef int          i32x4 __attribute__((ext_vector_type(4)));
typedef int          i32x8 __attribute__((ext_vector_type(8)));

// ---------------- workspace layout (bytes) ----------------
// XBF   @ 0          : NPTS*DDIM bf16          = 4,194,304
// MBF   @ 4194304    : KCMP*DDIM*DDIM bf16     = 8,388,608
// H     @ 12582912   : KCMP*DDIM f32           =    65,536
// CK    @ 12648448   : KCMP f32                =       256
// SCORE @ 12648704   : NPTS*KCMP f32           = 2,097,152

// ---------- kernel 1: fp32 -> bf16 conversion of X ----------
__global__ __launch_bounds__(256)
void gmm_cvt_kernel(const float* __restrict__ x, bf16_t* __restrict__ xb, int n) {
    int i = blockIdx.x * blockDim.x + threadIdx.x;
    if (i < n) xb[i] = (bf16_t)x[i];
}

// ---------- kernel 2: per-component prep (unchanged, proven) ----------
__global__ __launch_bounds__(256)
void gmm_prep_kernel(const float* __restrict__ scale_raw,
                     const float* __restrict__ means,
                     const float* __restrict__ wraw,
                     bf16_t* __restrict__ Mout,
                     float* __restrict__ hout,
                     float* __restrict__ ckout) {
    const int k = blockIdx.x;
    const int t = threadIdx.x;
    extern __shared__ char smem[];
    float*  rowbuf = (float*)smem;                          // 256 f32
    bf16_t* Mb     = (bf16_t*)(smem + 256 * sizeof(float)); // 256x256 bf16

    const float* Lk = scale_raw + (size_t)k * DDIM * DDIM;

    for (int i = 0; i < DDIM; ++i) {
        float v = Lk[(size_t)i * DDIM + t];
        rowbuf[t] = (t < i) ? v : 0.0f;
        __syncthreads();
        float dinv = 1.0f / expf(Lk[(size_t)i * DDIM + i]);
        float out = 0.0f;
        if (t < i) {
            float s = 0.0f;
            for (int p = t; p < i; ++p)
                s += rowbuf[p] * (float)Mb[p * DDIM + t];
            out = -dinv * s;
        } else if (t == i) {
            out = dinv;
        }
        Mb[i * DDIM + t] = (bf16_t)out;
        __syncthreads();
    }

    bf16_t* Mk = Mout + (size_t)k * DDIM * DDIM;
    for (int r = 0; r < DDIM; ++r)
        Mk[(size_t)r * DDIM + t] = Mb[r * DDIM + t];

    const float* mu = means + (size_t)k * DDIM;
    float hs = 0.0f;
    for (int j = 0; j <= t; ++j)
        hs += (float)Mb[t * DDIM + j] * mu[j];
    hout[(size_t)k * DDIM + t] = hs;

    __syncthreads();
    rowbuf[t] = Lk[(size_t)t * DDIM + t];
    __syncthreads();
    if (t == 0) {
        float c = 0.0f;
        for (int i = 0; i < DDIM; ++i) c += rowbuf[i];
        ckout[k] = wraw[k] - c;
    }
}

// ---------- TDM: DMA one 16x256 bf16 tile (row-major, 8 KB) into LDS ----------
// D# per CDNA5 ISA: group0 = {count/flags, lds_addr, gaddr_lo, gaddr_hi|type=2}
// group1: data_size=2B (code 1), tensor_dim0=256, tensor_dim1=16,
//         tile_dim0=256, tile_dim1=16, tensor_dim0_stride=256. 2D -> groups 2/3 zero.
// This toolchain carries the 6-arg builtin: (u32x4, i32x8, i32x4, i32x4, i32x8, cpol).
__device__ __forceinline__ void tdm_load_tile(const bf16_t* gsrc, unsigned int lds_off) {
    unsigned long long ga = (unsigned long long)(uintptr_t)gsrc;
    u32x4 g0;
    g0[0] = 1u;                                          // count=1, user mode
    g0[1] = lds_off;                                     // lds_addr (bytes)
    g0[2] = (unsigned int)ga;                            // global_addr[31:0]
    g0[3] = (unsigned int)(ga >> 32) | (2u << 30);       // global_addr[56:32] | type=2
    i32x8 g1;
    g1[0] = 0x00010000;   // workgroup_mask=0, data_size=1 (2 bytes)
    g1[1] = 0x01000000;   // tensor_dim0[15:0]=256 at bits [31:16]
    g1[2] = 0x00100000;   // tensor_dim0[31:16]=0, tensor_dim1[15:0]=16 at [31:16]
    g1[3] = 0x01000000;   // tensor_dim1[31:16]=0, tile_dim0=256 at [31:16]
    g1[4] = 16;           // tile_dim1=16, tile_dim2=0
    g1[5] = 256;          // tensor_dim0_stride[31:0]=256
    g1[6] = 0;            // stride0 hi, tensor_dim1_stride lo (unused for 2D tile)
    g1[7] = 0;
    i32x4 z4 = {0, 0, 0, 0};
    i32x8 z8 = {0, 0, 0, 0, 0, 0, 0, 0};
    __builtin_amdgcn_tensor_load_to_lds(g0, g1, z4, z4, z8, 0);
}

// ---------- kernel 3: fused WMMA GEMM + squared-residual reduce ----------
// grid = (NPTS/256, KCMP), block = 256 (8 wave32).
// Wave w holds TWO 16-row A strips in registers (rows m0..m0+31); the 16x256
// M_k tile for the current output-column block is TDM-staged into LDS once per
// workgroup (double-buffered) and shared by all 8 waves; each LDS B fragment
// feeds two WMMAs.
__global__ __launch_bounds__(256, 1)
void gmm_score_kernel(const bf16_t* __restrict__ Xb,
                      const bf16_t* __restrict__ Mbf,
                      const float* __restrict__ h,
                      const float* __restrict__ ck,
                      float* __restrict__ score) {
    __shared__ __align__(16) bf16_t sb[2][16 * DDIM];   // 2 x 8 KB

    const int k    = blockIdx.y;
    const int wave = threadIdx.x >> 5;
    const int lane = threadIdx.x & 31;
    const int m0   = blockIdx.x * 256 + wave * 32;
    const int half = (lane < 16) ? 0 : 1;
    const int l16  = lane & 15;

    const bf16_t* Mk = Mbf + (size_t)k * DDIM * DDIM;
    unsigned int sboff0 = (unsigned int)(uintptr_t)(&sb[0][0]);
    unsigned int sboff1 = (unsigned int)(uintptr_t)(&sb[1][0]);

    // A fragments: 16-bit A 16x32 layout (lane<16 -> K {0..7,16..23}; lane>=16 -> +8)
    const int klo = half ? 8 : 0;
    const bf16_t* xrow0 = Xb + (size_t)(m0 + l16) * DDIM;
    const bf16_t* xrow1 = Xb + (size_t)(m0 + 16 + l16) * DDIM;
    v16bf A0[8], A1[8];
#pragma unroll
    for (int s = 0; s < 8; ++s) {
        const bf16_t* p0 = xrow0 + s * 32 + klo;
        const bf16_t* p1 = xrow1 + s * 32 + klo;
        bf16x8 a0 = *(const bf16x8*)(p0);
        bf16x8 a1 = *(const bf16x8*)(p0 + 16);
        bf16x8 b0 = *(const bf16x8*)(p1);
        bf16x8 b1 = *(const bf16x8*)(p1 + 16);
        v16bf va, vb;
#pragma unroll
        for (int e = 0; e < 8; ++e) {
            va[e] = a0[e]; va[e + 8] = a1[e];
            vb[e] = b0[e]; vb[e + 8] = b1[e];
        }
        A0[s] = va;
        A1[s] = vb;
    }

    const float* hk = h + (size_t)k * DDIM;
    float acc0[8], acc1[8];
#pragma unroll
    for (int r = 0; r < 8; ++r) { acc0[r] = 0.0f; acc1[r] = 0.0f; }

    // prologue: DMA first M tile
    if (wave == 0) tdm_load_tile(Mk, sboff0);

    const int kbase = half ? 8 : 0;
    for (int ic = 0; ic < 16; ++ic) {
        if (wave == 0) {
            if (ic + 1 < 16) {
                tdm_load_tile(Mk + (size_t)(ic + 1) * 16 * DDIM,
                              ((ic + 1) & 1) ? sboff1 : sboff0);
                __builtin_amdgcn_s_wait_tensorcnt((short)1);  // current tile done
            } else {
                __builtin_amdgcn_s_wait_tensorcnt((short)0);
            }
        }
        __syncthreads();   // publish staged tile to all waves

        // B 32x16 layout (matches A): lane<16 -> K {0..7,16..23}; lane>=16 -> +8
        const bf16_t* bbase = &sb[ic & 1][l16 * DDIM + kbase];
        v8f c0 = {}, c1 = {};
#pragma unroll
        for (int s = 0; s < 8; ++s) {
            bf16x8 bl0 = *(const bf16x8*)(bbase + s * 32);
            bf16x8 bl1 = *(const bf16x8*)(bbase + s * 32 + 16);
            v16bf b;
#pragma unroll
            for (int e = 0; e < 8; ++e) { b[e] = bl0[e]; b[e + 8] = bl1[e]; }
            c0 = __builtin_amdgcn_wmma_f32_16x16x32_bf16(
                     false, A0[s], false, b, (short)0, c0, false, false);
            c1 = __builtin_amdgcn_wmma_f32_16x16x32_bf16(
                     false, A1[s], false, b, (short)0, c1, false, false);
        }
        float hv = hk[ic * 16 + l16];
#pragma unroll
        for (int r = 0; r < 8; ++r) {
            float d0 = c0[r] - hv; acc0[r] += d0 * d0;
            float d1 = c1[r] - hv; acc1[r] += d1 * d1;
        }
        __syncthreads();   // all reads done before next overwrite of this buffer
    }

    // reduce over the 16 lanes of each half -> per-row maha
#pragma unroll
    for (int m = 1; m < 16; m <<= 1) {
#pragma unroll
        for (int r = 0; r < 8; ++r) {
            acc0[r] += __shfl_xor(acc0[r], m, 32);
            acc1[r] += __shfl_xor(acc1[r], m, 32);
        }
    }

    if (l16 == 0) {
        float c0k = ck[k];
        int rb0 = m0 + (half ? 8 : 0);
        int rb1 = rb0 + 16;
#pragma unroll
        for (int r = 0; r < 8; ++r) {
            score[(size_t)(rb0 + r) * KCMP + k] = -0.5f * acc0[r] + c0k;
            score[(size_t)(rb1 + r) * KCMP + k] = -0.5f * acc1[r] + c0k;
        }
    }
}

// ---------- kernel 4: argmin over K (first-min tie-break) ----------
__global__ __launch_bounds__(256)
void gmm_argmin_kernel(const float* __restrict__ score, float* __restrict__ out) {
    int n = blockIdx.x * blockDim.x + threadIdx.x;
    if (n >= NPTS) return;
    const float* s = score + (size_t)n * KCMP;
    float best = s[0];
    int   bi   = 0;
    for (int k2 = 1; k2 < KCMP; ++k2) {
        float v = s[k2];
        if (v < best) { best = v; bi = k2; }
    }
    out[n] = (float)bi;
}

extern "C" void kernel_launch(void* const* d_in, const int* in_sizes, int n_in,
                              void* d_out, int out_size, void* d_ws, size_t ws_size,
                              hipStream_t stream) {
    const float* x         = (const float*)d_in[0]; // [N, D]
    const float* means     = (const float*)d_in[1]; // [K, D]
    const float* scale_raw = (const float*)d_in[2]; // [K, D, D]
    const float* wraw      = (const float*)d_in[3]; // [K]

    char* ws = (char*)d_ws;
    bf16_t* Xb    = (bf16_t*)(ws);
    bf16_t* Mbf   = (bf16_t*)(ws + 4194304);
    float*  hbuf  = (float*) (ws + 12582912);
    float*  ckbuf = (float*) (ws + 12648448);
    float*  score = (float*) (ws + 12648704);
    float*  out   = (float*)d_out;

    gmm_cvt_kernel<<<(NPTS * DDIM) / 256, 256, 0, stream>>>(x, Xb, NPTS * DDIM);

    size_t lds = 256 * sizeof(float) + (size_t)DDIM * DDIM * sizeof(bf16_t);
    gmm_prep_kernel<<<KCMP, 256, lds, stream>>>(scale_raw, means, wraw,
                                                Mbf, hbuf, ckbuf);

    gmm_score_kernel<<<dim3(NPTS / 256, KCMP), 256, 0, stream>>>(
        Xb, Mbf, hbuf, ckbuf, score);

    gmm_argmin_kernel<<<NPTS / 256, 256, 0, stream>>>(score, out);
}
